// GAT_48704929136998
// MI455X (gfx1250) — matrix operand order repaired and checked by reading, not compiled, not run on hardware
//
#include <hip/hip_runtime.h>
#include <math.h>

#define N_NODES 50000
#define N_EDGES 800000
#define NEG_SLOPE 0.2f

typedef __attribute__((ext_vector_type(16))) _Float16 v16h;
typedef __attribute__((ext_vector_type(8)))  _Float16 v8h;
typedef __attribute__((ext_vector_type(8)))  float    v8f;
typedef __attribute__((ext_vector_type(4)))  int      v4i;

typedef __attribute__((address_space(1))) v4i* as1_v4ip;
typedef __attribute__((address_space(3))) v4i* as3_v4ip;

#if defined(__has_builtin)
#if __has_builtin(__builtin_amdgcn_global_load_async_to_lds_b128)
#define USE_ASYNC_LDS 1
#else
#define USE_ASYNC_LDS 0
#endif
#else
#define USE_ASYNC_LDS 0
#endif

static inline int cdiv(int a, int b) { return (a + b - 1) / b; }

// Ordered-int trick for float atomic max (memory initialized to -inf-ish).
__device__ __forceinline__ void atomicMaxF(float* addr, float val) {
  int iv = __float_as_int(val);
  if (iv >= 0) atomicMax((int*)addr, iv);
  else         atomicMin((unsigned int*)addr, __float_as_uint(val));
}

// ---------------- conversion kernels ----------------
__global__ void k_f32_to_f16(const float* __restrict__ in, _Float16* __restrict__ out, int n) {
  int i = blockIdx.x * blockDim.x + threadIdx.x;
  if (i < n) out[i] = (_Float16)in[i];
}

// WT[j, i] = (f16) W[i, j] : makes the WMMA B operand contiguous along K.
__global__ void k_w_transpose_f16(const float* __restrict__ W, _Float16* __restrict__ WT,
                                  int K, int Kout) {
  int idx = blockIdx.x * blockDim.x + threadIdx.x;
  if (idx >= K * Kout) return;
  int i = idx / Kout;
  int j = idx - i * Kout;
  WT[(size_t)j * K + i] = (_Float16)W[idx];
}

// ---------------- WMMA GEMM (K fixed at 256) ----------------
// C[M,Kout](f32) = A[M,256](f16) * B, with WT[Kout,256] = B^T.
// Block = 8 waves. Block stages one 64-col x 256-K weight panel into LDS
// (async-to-LDS when available), then each wave computes a 16x64 strip:
// A tile loaded once per k-step, reused by 4 WMMAs reading B from LDS.
#define BS_STRIDE 264  // 256 + 8 halves pad -> lanes hit distinct LDS banks

__global__ __launch_bounds__(256)
void k_wmma_gemm(const _Float16* __restrict__ A, const _Float16* __restrict__ WT,
                 float* __restrict__ C, int M, int K, int Kout) {
  __shared__ _Float16 Bs[64 * BS_STRIDE];

  const int cg = blockIdx.y;  // 64-column group
  {  // stage WT rows [cg*64, cg*64+64) (each 256 halves) into padded LDS rows
    const _Float16* gbase = WT + (size_t)cg * 64 * 256;
    int t = threadIdx.x;
#pragma unroll
    for (int c = 0; c < 8; ++c) {
      int chunk = t + (c << 8);   // 0..2047 chunks of 16B (8 halves)
      int r   = chunk >> 5;       // row 0..63
      int k16 = chunk & 31;       // 16B chunk within row
      const _Float16* g = gbase + (size_t)r * 256 + k16 * 8;
      _Float16* l = &Bs[r * BS_STRIDE + k16 * 8];
#if USE_ASYNC_LDS
      __builtin_amdgcn_global_load_async_to_lds_b128(
          (as1_v4ip)(_Float16*)g, (as3_v4ip)l, 0, 0);
#else
      *(v8h*)l = *(const v8h*)g;
#endif
    }
#if USE_ASYNC_LDS
#if __has_builtin(__builtin_amdgcn_s_wait_asynccnt)
    __builtin_amdgcn_s_wait_asynccnt(0);
#else
    asm volatile("s_wait_asynccnt 0" ::: "memory");
#endif
#endif
  }
  __syncthreads();

  int wave = threadIdx.x >> 5;
  int tm = blockIdx.x * 8 + wave;
  if (tm >= (M >> 4)) return;            // wave-uniform; EXEC stays all-ones
  int lane = threadIdx.x & 31;
  int kg = lane >> 4;                    // K-group (0: low half, 1: high half)
  int mr = lane & 15;                    // A row / B col / C col within tile

  const _Float16* aRow = A + (size_t)(tm * 16 + mr) * K;

  v8f acc[4] = {};
#pragma unroll
  for (int kb = 0; kb < 256; kb += 32) {
    // A 16x32 f16 layout: kg=0 holds K {0..7,16..23}, kg=1 holds {8..15,24..31}
    v8h a0 = *(const v8h*)(aRow + kb + 8 * kg);
    v8h a1 = *(const v8h*)(aRow + kb + 16 + 8 * kg);
    v16h a;
#pragma unroll
    for (int i = 0; i < 8; ++i) { a[i] = a0[i]; a[i + 8] = a1[i]; }
#pragma unroll
    for (int j = 0; j < 4; ++j) {
      // B 32x16 layout: kg=0 holds K {0..7,16..23}, kg=1 holds {8..15,24..31}
      const _Float16* bb = &Bs[(j * 16 + mr) * BS_STRIDE + kb + 8 * kg];
      v8h b0 = *(const v8h*)(bb);
      v8h b1 = *(const v8h*)(bb + 16);
      v16h b;
#pragma unroll
      for (int i = 0; i < 8; ++i) { b[i] = b0[i]; b[i + 8] = b1[i]; }
      acc[j] = __builtin_amdgcn_wmma_f32_16x16x32_f16(false, a, false, b,
                                                      (short)0, acc[j], false, false);
    }
  }
#pragma unroll
  for (int j = 0; j < 4; ++j) {
    float* cp = C + (size_t)(tm * 16 + 8 * kg) * Kout + cg * 64 + j * 16 + mr;
#pragma unroll
    for (int r = 0; r < 8; ++r) cp[(size_t)r * Kout] = acc[j][r];
  }
}

// ---------------- attention coefficients ----------------
template <int H>
__global__ void k_eler(const float* __restrict__ feat, const float* __restrict__ al,
                       const float* __restrict__ ar, float* __restrict__ el,
                       float* __restrict__ er, int N) {
  int idx = blockIdx.x * blockDim.x + threadIdx.x;
  if (idx >= N * H) return;
  int n = idx / H;
  int h = idx - n * H;
  const float* f = feat + (size_t)n * (H * 64) + h * 64;
  float sl = 0.f, sr = 0.f;
#pragma unroll
  for (int d = 0; d < 64; ++d) {
    float v = f[d];
    sl += v * al[h * 64 + d];
    sr += v * ar[h * 64 + d];
  }
  el[idx] = sl;
  er[idx] = sr;
}

// ---------------- init kernels ----------------
__global__ void k_init_softmax(float* __restrict__ emax, float* __restrict__ denom, int n) {
  int i = blockIdx.x * blockDim.x + threadIdx.x;
  if (i < n) { emax[i] = -3.0e38f; denom[i] = 0.f; }
}

template <int H>
__global__ void k_init_bias(float* __restrict__ out, const float* __restrict__ bias, int total) {
  const int HD = H * 64;
  int i = blockIdx.x * blockDim.x + threadIdx.x;
  if (i < total) out[i] = bias[i % HD];
}

template <int H>
__global__ void k_init_bias_res(float* __restrict__ out, const float* __restrict__ bias,
                                const float* __restrict__ res, int total) {
  const int HD = H * 64;
  int i = blockIdx.x * blockDim.x + threadIdx.x;
  if (i < total) out[i] = bias[i % HD] + res[i];
}

// ---------------- edge passes (segment softmax + aggregate) ----------------
template <int H>
__global__ void k_edge_logits(const int* __restrict__ src, const int* __restrict__ dst,
                              const float* __restrict__ el, const float* __restrict__ er,
                              float* __restrict__ alpha, float* __restrict__ emax, int E) {
  int idx = blockIdx.x * blockDim.x + threadIdx.x;
  if (idx >= E * H) return;
  int e = idx / H;
  int h = idx - e * H;
  int s = src[e], d = dst[e];
  float v = el[s * H + h] + er[d * H + h];
  v = (v > 0.f) ? v : NEG_SLOPE * v;
  alpha[idx] = v;
  atomicMaxF(&emax[d * H + h], v);
}

template <int H>
__global__ void k_edge_exp(const int* __restrict__ dst, float* __restrict__ alpha,
                           const float* __restrict__ emax, float* __restrict__ denom, int E) {
  int idx = blockIdx.x * blockDim.x + threadIdx.x;
  if (idx >= E * H) return;
  int e = idx / H;
  int h = idx - e * H;
  int d = dst[e];
  float ee = __expf(alpha[idx] - emax[d * H + h]);
  alpha[idx] = ee;
  atomicAdd(&denom[d * H + h], ee);
}

template <int H>
__global__ void k_edge_norm(const int* __restrict__ dst, float* __restrict__ alpha,
                            const float* __restrict__ denom, int E) {
  int idx = blockIdx.x * blockDim.x + threadIdx.x;
  if (idx >= E * H) return;
  int e = idx / H;
  int h = idx - e * H;
  alpha[idx] = alpha[idx] / denom[dst[e] * H + h];
}

// One wave per edge; lanes stride the H*64 feature dims.
template <int H>
__global__ void k_edge_aggregate(const int* __restrict__ src, const int* __restrict__ dst,
                                 const float* __restrict__ alpha, const float* __restrict__ feat,
                                 float* __restrict__ out, int E) {
  int e = blockIdx.x * (blockDim.x >> 5) + (threadIdx.x >> 5);
  if (e >= E) return;
  int lane = threadIdx.x & 31;
  int s = src[e], d = dst[e];
  const int HD = H * 64;
  const float* fs = feat + (size_t)s * HD;
  float* od = out + (size_t)d * HD;
  __builtin_prefetch(fs, 0, 0);  // global_prefetch_b8: warm the gathered row
#pragma unroll
  for (int i = lane; i < HD; i += 32) {
    float a = alpha[(size_t)e * H + (i >> 6)];
    atomicAdd(&od[i], fs[i] * a);
  }
}

__global__ void k_elu(float* __restrict__ x, int n) {
  int i = blockIdx.x * blockDim.x + threadIdx.x;
  if (i < n) {
    float v = x[i];
    x[i] = (v > 0.f) ? v : expm1f(v);
  }
}

// ---------------- per-layer driver ----------------
template <int H>
static void run_gat_layer(const float* hIn, const int* src, const int* dst,
                          const float* W, const float* al, const float* ar, const float* bias,
                          const float* resAdd, const float* resW, float* resBuf,
                          bool act, float* outAgg,
                          float* featF, float* el, float* er, float* emax, float* denom,
                          float* alpha, _Float16* h16, _Float16* w16, hipStream_t stream) {
  const int N = N_NODES, E = N_EDGES, K = 256, Kout = H * 64;
  k_f32_to_f16<<<cdiv(N * K, 256), 256, 0, stream>>>(hIn, h16, N * K);
  k_w_transpose_f16<<<cdiv(K * Kout, 256), 256, 0, stream>>>(W, w16, K, Kout);
  dim3 gg(cdiv(N / 16, 8), Kout / 64);
  k_wmma_gemm<<<gg, 256, 0, stream>>>(h16, w16, featF, N, K, Kout);

  const float* resPtr = resAdd;
  if (resW) {  // linear residual (layer 2): reuse h16, overwrite w16 after main GEMM
    k_w_transpose_f16<<<cdiv(K * Kout, 256), 256, 0, stream>>>(resW, w16, K, Kout);
    k_wmma_gemm<<<gg, 256, 0, stream>>>(h16, w16, resBuf, N, K, Kout);
    resPtr = resBuf;
  }

  k_eler<H><<<cdiv(N * H, 256), 256, 0, stream>>>(featF, al, ar, el, er, N);
  k_init_softmax<<<cdiv(N * H, 256), 256, 0, stream>>>(emax, denom, N * H);
  if (resPtr)
    k_init_bias_res<H><<<cdiv(N * Kout, 256), 256, 0, stream>>>(outAgg, bias, resPtr, N * Kout);
  else
    k_init_bias<H><<<cdiv(N * Kout, 256), 256, 0, stream>>>(outAgg, bias, N * Kout);

  k_edge_logits<H><<<cdiv(E * H, 256), 256, 0, stream>>>(src, dst, el, er, alpha, emax, E);
  k_edge_exp<H><<<cdiv(E * H, 256), 256, 0, stream>>>(dst, alpha, emax, denom, E);
  k_edge_norm<H><<<cdiv(E * H, 256), 256, 0, stream>>>(dst, alpha, denom, E);
  k_edge_aggregate<H><<<cdiv(E, 8), 256, 0, stream>>>(src, dst, alpha, featF, outAgg, E);
  if (act) k_elu<<<cdiv(N * Kout, 256), 256, 0, stream>>>(outAgg, N * Kout);
}

extern "C" void kernel_launch(void* const* d_in, const int* in_sizes, int n_in,
                              void* d_out, int out_size, void* d_ws, size_t ws_size,
                              hipStream_t stream) {
  (void)in_sizes; (void)n_in; (void)out_size; (void)ws_size;
  const float* x   = (const float*)d_in[0];
  const int*   src = (const int*)d_in[1];
  const int*   dst = (const int*)d_in[2];
  const float* W0  = (const float*)d_in[3];
  const float* al0 = (const float*)d_in[4];
  const float* ar0 = (const float*)d_in[5];
  const float* b0  = (const float*)d_in[6];
  const float* W1  = (const float*)d_in[7];
  const float* al1 = (const float*)d_in[8];
  const float* ar1 = (const float*)d_in[9];
  const float* b1  = (const float*)d_in[10];
  const float* W2  = (const float*)d_in[11];
  const float* al2 = (const float*)d_in[12];
  const float* ar2 = (const float*)d_in[13];
  const float* b2  = (const float*)d_in[14];
  const float* rw2 = (const float*)d_in[15];

  const size_t N = N_NODES, E = N_EDGES;
  // workspace carve-out (~208 MB)
  float* featF = (float*)d_ws;                 // [N,256] GEMM output / gather source
  float* aggA  = featF + N * 256;              // [N,256] layer-0 output (h1)
  float* aggB  = aggA + N * 256;               // [N,256] layer-1 output (h2)
  float* el    = aggB + N * 256;               // [N,4]
  float* er    = el + N * 4;                   // [N,4]
  float* emax  = er + N * 4;                   // [N,4]
  float* denom = emax + N * 4;                 // [N,4]
  float* alpha = denom + N * 4;                // [E,4]
  float* resb  = alpha + E * 4;                // [N,64] layer-2 residual features
  _Float16* h16 = (_Float16*)(resb + N * 64);  // [N,256] f16 node features
  _Float16* w16 = h16 + N * 256;               // [256,256] f16 transposed weights

  // layer 0: 256 -> 4x64, no residual, ELU
  run_gat_layer<4>(x, src, dst, W0, al0, ar0, b0, nullptr, nullptr, nullptr,
                   true, aggA, featF, el, er, emax, denom, alpha, h16, w16, stream);
  // layer 1: 256 -> 4x64, identity residual, ELU
  run_gat_layer<4>(aggA, src, dst, W1, al1, ar1, b1, aggA, nullptr, nullptr,
                   true, aggB, featF, el, er, emax, denom, alpha, h16, w16, stream);
  // layer 2: 256 -> 1x64, linear residual (res_w2), no activation -> d_out
  run_gat_layer<1>(aggB, src, dst, W2, al2, ar2, b2, nullptr, rw2, resb,
                   false, (float*)d_out, featF, el, er, emax, denom, alpha, h16, w16, stream);
}
